// MultiheadAttention_386547057425
// MI455X (gfx1250) — matrix operand order repaired and checked by reading, not compiled, not run on hardware
//
#include <hip/hip_runtime.h>
#include <hip/hip_bf16.h>

// ---------------------------------------------------------------------------
// CDNA5 (gfx1250) multi-head attention: bf16 WMMA everywhere, async-LDS
// double buffering for the flash-attention K/V stream.
//   B=2, S=4096, D=512, H=8, hd=64.  All GEMMs via v_wmma_f32_16x16x32_bf16.
// ---------------------------------------------------------------------------

typedef __attribute__((ext_vector_type(16))) __bf16 bf16x16;
typedef __attribute__((ext_vector_type(8)))  __bf16 bf16x8;
typedef __attribute__((ext_vector_type(8)))  float  f32x8;

static __device__ __forceinline__ unsigned lane_id() { return threadIdx.x & 31u; }

static __device__ __forceinline__ __bf16 f2bf(float f) {
  union { float f; unsigned u; } a; a.f = f;
  unsigned r = a.u + 0x7fffu + ((a.u >> 16) & 1u);   // round-to-nearest-even
  union { unsigned short s; __bf16 b; } o; o.s = (unsigned short)(r >> 16);
  return o.b;
}

// A-matrix fragment (16x32 bf16, row-major, stride lda elements).
// ISA layout: lane L holds row M=L&15; VGPR0-3 -> K = 8*hi+0..7, VGPR4-7 -> K = 16+8*hi+0..7.
static __device__ __forceinline__ bf16x16 load_a_frag(const __bf16* A, int lda) {
  unsigned l = lane_id();
  const __bf16* p = A + (int)(l & 15u) * lda + ((l >> 4) ? 8 : 0);
  union { bf16x16 v; struct { bf16x8 lo, hi; } s; } u;
  u.s.lo = *(const bf16x8*)(p);
  u.s.hi = *(const bf16x8*)(p + 16);
  return u.v;
}

// B-matrix fragment (KxN = 32x16).  Source W is [N,K] row-major (stride ldw):
// B[k][n] = W[n][k].  ISA layout: lane holds N = lane&15; VGPR0-3 -> K = 8*hi+0..7, VGPR4-7 -> K = 16+8*hi+0..7.
static __device__ __forceinline__ bf16x16 load_b_frag(const __bf16* W, int ldw) {
  unsigned l = lane_id();
  const __bf16* p = W + (int)(l & 15u) * ldw + ((l >> 4) ? 8 : 0);
  union { bf16x16 v; struct { bf16x8 lo, hi; } s; } u;
  u.s.lo = *(const bf16x8*)(p);
  u.s.hi = *(const bf16x8*)(p + 16);
  return u.v;
}

static __device__ __forceinline__ f32x8 wmma_bf16(bf16x16 a, bf16x16 b, f32x8 c) {
  return __builtin_amdgcn_wmma_f32_16x16x32_bf16(false, a, false, b, (short)0, c,
                                                 false, false);
}

static __device__ __forceinline__ float rowred_max(float v) {
  #pragma unroll
  for (int m = 1; m < 16; m <<= 1) v = fmaxf(v, __shfl_xor(v, m, 32));
  return v;
}
static __device__ __forceinline__ float rowred_sum(float v) {
  #pragma unroll
  for (int m = 1; m < 16; m <<= 1) v += __shfl_xor(v, m, 32);
  return v;
}

// Async global->LDS copy of 16 bytes per lane (GLOBAL_LOAD_ASYNC_TO_LDS_B128,
// tracked by ASYNCcnt).  Inline asm: portable across ROCm 7.2 / amdgpu-toolchain.
static __device__ __forceinline__ void async_ld16(__bf16* lds, const __bf16* g) {
  unsigned loff =
      (unsigned)(size_t)(__attribute__((address_space(3))) __bf16*)lds;
  asm volatile("global_load_async_to_lds_b128 %0, %1, off"
               :: "v"(loff), "v"((unsigned long long)(size_t)g)
               : "memory");
}
static __device__ __forceinline__ void wait_async0() {
  asm volatile("s_wait_asynccnt 0x0" ::: "memory");
}

// ---------------------------------------------------------------------------
// Kernel 1: f32 -> bf16 conversion
// ---------------------------------------------------------------------------
__global__ void cvt_f32_bf16(const float* __restrict__ in, __bf16* __restrict__ out,
                             int n) {
  int i = blockIdx.x * blockDim.x + threadIdx.x;
  if (i < n) out[i] = f2bf(in[i]);
}

// ---------------------------------------------------------------------------
// Kernel 2: QKV projection  C[8192,1536] = X[8192,512] @ Wqkv[1536,512]^T + b
// Each wave computes a 16x64 tile.  Epilogue scatters into per-head
// Q [16,4096,64], K [16,4096,64] and V^T [16,64,4096] bf16 buffers.
// ---------------------------------------------------------------------------
__global__ void __launch_bounds__(256) qkv_gemm(
    const __bf16* __restrict__ X, const __bf16* __restrict__ Wq,
    const float* __restrict__ bias, __bf16* __restrict__ Qb,
    __bf16* __restrict__ Kb, __bf16* __restrict__ Vt) {
  const int Kd = 512, MT = 8192 / 16;
  int wave = blockIdx.x * (blockDim.x >> 5) + (threadIdx.x >> 5);
  int mt = wave % MT, ng = wave / MT;       // ng in [0,24)
  int m0 = mt * 16, n0 = ng * 64;
  const __bf16* Arow = X + m0 * Kd;
  f32x8 acc[4] = {};
  for (int k0 = 0; k0 < Kd; k0 += 32) {
    __builtin_prefetch(Arow + k0 + 64, 0, 1);
    bf16x16 a = load_a_frag(Arow + k0, Kd);
    #pragma unroll
    for (int t = 0; t < 4; ++t)
      acc[t] = wmma_bf16(a, load_b_frag(Wq + (n0 + 16 * t) * Kd + k0, Kd), acc[t]);
  }
  unsigned l = lane_id();
  int hi = l >> 4, cn = l & 15;
  #pragma unroll
  for (int t = 0; t < 4; ++t) {
    int e = n0 + 16 * t + cn;
    int h = e / 192, sub = e % 192, sel = sub >> 6, d = sub & 63;
    float bv = bias[e];
    #pragma unroll
    for (int r = 0; r < 8; ++r) {
      int gm = m0 + r + 8 * hi;
      int bb = gm >> 12, s = gm & 4095;
      __bf16 v = f2bf(acc[t][r] + bv);
      int head = bb * 8 + h;
      if (sel == 0)      Qb[(head * 4096 + s) * 64 + d] = v;
      else if (sel == 1) Kb[(head * 4096 + s) * 64 + d] = v;
      else               Vt[(head * 64 + d) * 4096 + s] = v;
    }
  }
}

// ---------------------------------------------------------------------------
// Kernel 3: flash attention.  One wave = one (b,h) head and one 16-query tile.
// K/V tiles (32 keys) are double-buffered in LDS via global_load_async_to_lds
// (ASYNCcnt pipelining): s_wait_asynccnt 0 for tile kt, then issue tile kt+1, feed
// the 8 WMMAs per step from ds_loads.  Online softmax uses shfl_xor row
// reductions on the C layout; P staged through per-wave LDS (C- -> A-layout).
// ---------------------------------------------------------------------------
__global__ void __launch_bounds__(64) flash_attn(
    const __bf16* __restrict__ Qb, const __bf16* __restrict__ Kb,
    const __bf16* __restrict__ Vt, __bf16* __restrict__ Ob) {
  __shared__ __align__(64) __bf16 Kls[2][2][32 * 64];  // [wave][buf][key][d]
  __shared__ __align__(64) __bf16 Vls[2][2][64 * 32];  // [wave][buf][d][key]
  __shared__ __align__(64) __bf16 tileP[2][16 * 32];
  const int S = 4096, NT = S / 32;
  int w = threadIdx.x >> 5;
  int wave = blockIdx.x * 2 + w;
  int bh = wave >> 8;           // 16 (b,h) heads
  int qt = wave & 255;          // 256 query tiles of 16
  const __bf16* Qt = Qb + (bh * S + qt * 16) * 64;
  const __bf16* Kh = Kb + bh * S * 64;
  const __bf16* Vh = Vt + bh * 64 * S;
  unsigned l = lane_id();
  int hi = l >> 4, cn = l & 15;

  // Async stage of one 32-key K tile (32x64) + V^T tile (64x32) into LDS buf.
  int krow = l >> 3, kch = (l & 7) * 8;   // 4 rows/instr * 8 iters = 32 rows
  int vrow = l >> 2, vch = (l & 3) * 8;   // 8 rows/instr * 8 iters = 64 rows
  auto stage = [&](int buf, int kt2) {
    const __bf16* Kg = Kh + kt2 * 32 * 64;
    const __bf16* Vg = Vh + kt2 * 32;
    __bf16* kb = Kls[w][buf];
    __bf16* vb = Vls[w][buf];
    #pragma unroll
    for (int i = 0; i < 8; ++i)
      async_ld16(kb + (krow + 4 * i) * 64 + kch, Kg + (krow + 4 * i) * 64 + kch);
    #pragma unroll
    for (int i = 0; i < 8; ++i)
      async_ld16(vb + (vrow + 8 * i) * 32 + vch, Vg + (vrow + 8 * i) * S + vch);
  };

  bf16x16 aq0 = load_a_frag(Qt, 64);        // d = 0..31
  bf16x16 aq1 = load_a_frag(Qt + 32, 64);   // d = 32..63
  f32x8 acc[4] = {};
  float mrun[8], lrun[8];
  #pragma unroll
  for (int r = 0; r < 8; ++r) { mrun[r] = -3.0e30f; lrun[r] = 0.f; }
  __bf16* tp = tileP[w];

  stage(0, 0);                              // prime the pipeline
  for (int kt = 0; kt < NT; ++kt) {
    int cur = kt & 1;
    wait_async0();                          // buf[cur] resident in LDS
    int nxt = (kt + 1 < NT) ? kt + 1 : kt;
    stage(cur ^ 1, nxt);                    // overlap next tile with compute
    const __bf16* kbc = Kls[w][cur];
    const __bf16* vbc = Vls[w][cur];

    f32x8 sa = {}, sb = {};
    sa = wmma_bf16(aq0, load_b_frag(kbc, 64), sa);            // keys 0..15, d 0..31
    sa = wmma_bf16(aq1, load_b_frag(kbc + 32, 64), sa);       // keys 0..15, d 32..63
    sb = wmma_bf16(aq0, load_b_frag(kbc + 16 * 64, 64), sb);  // keys 16..31
    sb = wmma_bf16(aq1, load_b_frag(kbc + 16 * 64 + 32, 64), sb);
    float alpha[8];
    #pragma unroll
    for (int r = 0; r < 8; ++r) {
      float xa = sa[r] * 0.125f;            // 1/sqrt(64)
      float xb = sb[r] * 0.125f;
      float rm = fmaxf(rowred_max(xa), rowred_max(xb));
      float mn = fmaxf(mrun[r], rm);
      alpha[r] = __expf(mrun[r] - mn);
      mrun[r] = mn;
      float pa = __expf(xa - mn);
      float pb = __expf(xb - mn);
      lrun[r] = lrun[r] * alpha[r] + rowred_sum(pa) + rowred_sum(pb);
      tp[(8 * hi + r) * 32 + cn]      = f2bf(pa);   // C-layout -> LDS
      tp[(8 * hi + r) * 32 + 16 + cn] = f2bf(pb);
    }
    #pragma unroll
    for (int t = 0; t < 4; ++t)
      #pragma unroll
      for (int r = 0; r < 8; ++r) acc[t][r] *= alpha[r];
    bf16x16 ap = load_a_frag(tp, 32);       // P as A-fragment (intra-wave DS order)
    #pragma unroll
    for (int t = 0; t < 4; ++t)             // V^T rows are d-columns in LDS
      acc[t] = wmma_bf16(ap, load_b_frag(vbc + (16 * t) * 32, 32), acc[t]);
  }
  wait_async0();                            // drain trailing prefetch

  int bb = bh >> 3, h = bh & 7;
  #pragma unroll
  for (int r = 0; r < 8; ++r) {
    float inv = 1.0f / lrun[r];
    int s = qt * 16 + r + 8 * hi;
    __bf16* orow = Ob + (bb * S + s) * 512 + h * 64;
    #pragma unroll
    for (int t = 0; t < 4; ++t) orow[16 * t + cn] = f2bf(acc[t][r] * inv);
  }
}

// ---------------------------------------------------------------------------
// Kernel 4: output projection  Out[8192,512] = vals[8192,512] @ o_w[512,512]^T + b
// ---------------------------------------------------------------------------
__global__ void __launch_bounds__(256) out_gemm(
    const __bf16* __restrict__ A, const __bf16* __restrict__ W,
    const float* __restrict__ bias, float* __restrict__ Out) {
  const int Kd = 512, MT = 8192 / 16;
  int wave = blockIdx.x * (blockDim.x >> 5) + (threadIdx.x >> 5);
  int mt = wave % MT, ng = wave / MT;       // ng in [0,8)
  int m0 = mt * 16, n0 = ng * 64;
  const __bf16* Ar = A + m0 * Kd;
  f32x8 acc[4] = {};
  for (int k0 = 0; k0 < Kd; k0 += 32) {
    __builtin_prefetch(Ar + k0 + 64, 0, 1);
    bf16x16 a = load_a_frag(Ar + k0, Kd);
    #pragma unroll
    for (int t = 0; t < 4; ++t)
      acc[t] = wmma_bf16(a, load_b_frag(W + (n0 + 16 * t) * Kd + k0, Kd), acc[t]);
  }
  unsigned l = lane_id();
  int hi = l >> 4, cn = l & 15;
  #pragma unroll
  for (int t = 0; t < 4; ++t) {
    int e = n0 + 16 * t + cn;
    float bv = bias[e];
    #pragma unroll
    for (int r = 0; r < 8; ++r)
      Out[(m0 + r + 8 * hi) * 512 + e] = acc[t][r] + bv;
  }
}

// ---------------------------------------------------------------------------
// Host side
// ---------------------------------------------------------------------------
extern "C" void kernel_launch(void* const* d_in, const int* in_sizes, int n_in,
                              void* d_out, int out_size, void* d_ws, size_t ws_size,
                              hipStream_t stream) {
  (void)in_sizes; (void)n_in; (void)out_size; (void)ws_size;
  const float* x     = (const float*)d_in[0];   // [2,4096,512]
  const float* qkv_w = (const float*)d_in[1];   // [1536,512]
  const float* qkv_b = (const float*)d_in[2];   // [1536]
  const float* o_w   = (const float*)d_in[3];   // [512,512]
  const float* o_b   = (const float*)d_in[4];   // [512]
  float* out = (float*)d_out;                   // [2,4096,512]

  char* w = (char*)d_ws;
  __bf16* x_bf   = (__bf16*)(w + 0);           // 8192*512
  __bf16* qw_bf  = (__bf16*)(w + 8388608);     // 1536*512
  __bf16* ow_bf  = (__bf16*)(w + 9961472);     // 512*512
  __bf16* q_bf   = (__bf16*)(w + 10485760);    // 16*4096*64
  __bf16* k_bf   = (__bf16*)(w + 18874368);    // 16*4096*64
  __bf16* vt_bf  = (__bf16*)(w + 27262976);    // 16*64*4096
  __bf16* val_bf = (__bf16*)(w + 35651584);    // 8192*512   (total ~42 MB)

  cvt_f32_bf16<<<(8192 * 512) / 256, 256, 0, stream>>>(x, x_bf, 8192 * 512);
  cvt_f32_bf16<<<(1536 * 512) / 256, 256, 0, stream>>>(qkv_w, qw_bf, 1536 * 512);
  cvt_f32_bf16<<<(512 * 512) / 256, 256, 0, stream>>>(o_w, ow_bf, 512 * 512);

  // 512 m-tiles * 24 n-groups = 12288 waves, 8 waves/block
  qkv_gemm<<<1536, 256, 0, stream>>>(x_bf, qw_bf, qkv_b, q_bf, k_bf, vt_bf);

  // 16 heads * 256 q-tiles = 4096 waves, 2 waves/block (34 KB LDS/block)
  flash_attn<<<2048, 64, 0, stream>>>(q_bf, k_bf, vt_bf, val_bf);

  // 512 m-tiles * 8 n-groups = 4096 waves, 8 waves/block
  out_gemm<<<512, 256, 0, stream>>>(val_bf, ow_bf, o_b, out);
}
